// HybridDecoderLayers_82540681494576
// MI455X (gfx1250) — matrix operand order repaired and checked by reading, compiled, not run on hardware
//
#include <hip/hip_runtime.h>
#include <cstddef>
#include <cstdint>

// ---------------------------------------------------------------- types
typedef __attribute__((ext_vector_type(16))) __bf16 v16bf;
typedef __attribute__((ext_vector_type(8)))  float  v8f;

// ---------------------------------------------------------------- model dims
#define TOK   8192          // b*l = 2*4096
#define BATCH 2
#define SEQ   4096
#define DM    2048
#define DI    4096
#define DXB   512
#define NH    64
#define HD    64
#define DSTATE 64
#define NC    64            // chunks per sequence
#define CHUNKL 64
#define INTER 5632
#define CDIM  5120          // 2*DXB + DI
#define NZ    9280          // DI + CDIM + NH
#define EPSF  1e-5f

// ---------------------------------------------------------------- helpers
__device__ __forceinline__ float siluf(float x){ return x / (1.f + __expf(-x)); }

__device__ __forceinline__ v8f wmma_bf(v16bf a, v16bf b, v8f c){
    return __builtin_amdgcn_wmma_f32_16x16x32_bf16(false, a, false, b, (short)0, c, false, false);
}

// CDNA5 async DMA: copy 16 bytes global -> LDS, tracked by ASYNCcnt (no VGPR data path)
__device__ __forceinline__ void async_ld16(uint32_t lds_off, const void* gaddr){
    asm volatile("global_load_async_to_lds_b128 %0, %1, off"
                 :: "v"(lds_off), "v"(gaddr) : "memory");
}
__device__ __forceinline__ void wait_async8(){
    asm volatile("s_wait_asynccnt 0x8" ::: "memory");
}
__device__ __forceinline__ void wait_async0(){
    asm volatile("s_wait_asynccnt 0x0" ::: "memory");
}

// A fragment (16x32, 16-bit): lane holds row M=lane&15; K = 16*(lane>>4) + element (contiguous)
__device__ __forceinline__ v16bf load_A_frag(const __bf16* a, int ld){
    int lane = threadIdx.x & 31;
    int row  = lane & 15;
    int kb   = (lane >> 4) * 16;
    v16bf f;
#pragma unroll
    for (int i = 0; i < 8; ++i){
        int k = kb + 2 * i;
        f[2*i]   = a[row*ld + k];
        f[2*i+1] = a[row*ld + k + 1];
    }
    return f;
}
// B fragment (32x16, 16-bit), source laid out [k][col] with stride ld
__device__ __forceinline__ v16bf load_B_frag(const __bf16* b, int ld){
    int lane = threadIdx.x & 31;
    int col  = lane & 15;
    int kb   = (lane >> 4) * 16;
    v16bf f;
#pragma unroll
    for (int v = 0; v < 8; ++v){
        int k = kb + 2*v;
        f[2*v]   = b[k*ld + col];
        f[2*v+1] = b[(k+1)*ld + col];
    }
    return f;
}
// B fragment where the source is stored transposed: element(k,col) = src[col*ld + k]
__device__ __forceinline__ v16bf load_Bt_frag(const __bf16* b, int ld){
    int lane = threadIdx.x & 31;
    int col  = lane & 15;
    int kb   = (lane >> 4) * 16;
    v16bf f;
#pragma unroll
    for (int v = 0; v < 8; ++v){
        int k = kb + 2*v;
        f[2*v]   = b[col*ld + k];
        f[2*v+1] = b[col*ld + k + 1];
    }
    return f;
}

// ---------------------------------------------------------------- fp32 -> bf16 hi/lo split
__global__ __launch_bounds__(256) void hdl_split(const float* __restrict__ w,
                                                 __bf16* __restrict__ hi,
                                                 __bf16* __restrict__ lo, size_t n){
    size_t i = (size_t)blockIdx.x * 256 + threadIdx.x;
    if (i < n){
        float v = w[i];
        __bf16 h = (__bf16)v;
        hi[i] = h;
        lo[i] = (__bf16)(v - (float)h);
    }
}

// ---------------------------------------------------------------- rmsnorm -> bf16 hi/lo
__global__ __launch_bounds__(256) void hdl_rmsnorm_split(const float* __restrict__ x,
                                                         const float* __restrict__ w,
                                                         __bf16* __restrict__ hi,
                                                         __bf16* __restrict__ lo, int D){
    int row = blockIdx.x;
    const float* xr = x + (size_t)row * D;
    __shared__ float red[256];
    float s = 0.f;
    for (int i = threadIdx.x; i < D; i += 256){ float v = xr[i]; s += v*v; }
    red[threadIdx.x] = s; __syncthreads();
    for (int o = 128; o > 0; o >>= 1){
        if ((int)threadIdx.x < o) red[threadIdx.x] += red[threadIdx.x + o];
        __syncthreads();
    }
    float inv = rsqrtf(red[0] / (float)D + EPSF);
    for (int i = threadIdx.x; i < D; i += 256){
        float v = xr[i] * inv * w[i];
        __bf16 h = (__bf16)v;
        hi[(size_t)row*D + i] = h;
        lo[(size_t)row*D + i] = (__bf16)(v - (float)h);
    }
}

// ---------------------------------------------------------------- gated rmsnorm (u = y*silu(z))
__global__ __launch_bounds__(256) void hdl_gatednorm_split(const float* __restrict__ Y,
                                                           const float* __restrict__ zxb,
                                                           const float* __restrict__ w,
                                                           __bf16* __restrict__ hi,
                                                           __bf16* __restrict__ lo){
    int row = blockIdx.x;
    const float* yr = Y   + (size_t)row * DI;
    const float* zr = zxb + (size_t)row * NZ;
    __shared__ float red[256];
    float loc[16]; float s = 0.f;
#pragma unroll
    for (int i = 0; i < 16; ++i){
        int c = threadIdx.x + i*256;
        float z = zr[c];
        float u = yr[c] * siluf(z);
        loc[i] = u; s += u*u;
    }
    red[threadIdx.x] = s; __syncthreads();
    for (int o = 128; o > 0; o >>= 1){
        if ((int)threadIdx.x < o) red[threadIdx.x] += red[threadIdx.x + o];
        __syncthreads();
    }
    float inv = rsqrtf(red[0] / (float)DI + EPSF);
#pragma unroll
    for (int i = 0; i < 16; ++i){
        int c = threadIdx.x + i*256;
        float v = loc[i] * inv * w[c];
        __bf16 h = (__bf16)v;
        hi[(size_t)row*DI + c] = h;
        lo[(size_t)row*DI + c] = (__bf16)(v - (float)h);
    }
}

// ---------------------------------------------------------------- depthwise causal conv + silu
__global__ __launch_bounds__(256) void hdl_conv_silu(const float* __restrict__ zxb,
                                                     const float* __restrict__ cw,
                                                     const float* __restrict__ cb,
                                                     float* __restrict__ xconv){
    size_t idx = (size_t)blockIdx.x * 256 + threadIdx.x;
    if (idx >= (size_t)TOK * CDIM) return;
    int ch = (int)(idx % CDIM);
    size_t t = idx / CDIM;
    int b = (int)(t / SEQ), l = (int)(t % SEQ);
    float acc = cb[ch];
#pragma unroll
    for (int j = 0; j < 4; ++j){
        int li = l - 3 + j;
        if (li >= 0)
            acc += zxb[((size_t)(b*SEQ + li))*NZ + DI + ch] * cw[ch*4 + j];
    }
    xconv[idx] = siluf(acc);
}

// ---------------------------------------------------------------- dt softplus, dA
__global__ __launch_bounds__(256) void hdl_dtprep(const float* __restrict__ zxb,
                                                  const float* __restrict__ dtb,
                                                  const float* __restrict__ Alog,
                                                  float* __restrict__ dtsp,
                                                  float* __restrict__ dAa){
    size_t idx = (size_t)blockIdx.x * 256 + threadIdx.x;
    if (idx >= (size_t)TOK * NH) return;
    int h = (int)(idx & 63);
    size_t t = idx >> 6;
    float x = zxb[t*NZ + (DI + CDIM) + h] + dtb[h];
    float sp = (x > 20.f) ? x : log1pf(__expf(x));
    dtsp[idx] = sp;
    dAa[idx]  = -__expf(Alog[h]) * sp;
}

// ---------------------------------------------------------------- per-(b,chunk,head) intra-chunk WMMA
__global__ __launch_bounds__(256) void hdl_chunk_fwd(const float* __restrict__ xconv,
                                                     const float* __restrict__ dtsp,
                                                     const float* __restrict__ dAa,
                                                     float* __restrict__ gAcum,
                                                     float* __restrict__ gStates,
                                                     float* __restrict__ gY){
    int blk = blockIdx.x;
    int h = blk & 63, c = (blk >> 6) & 63, b = blk >> 12;
    int g = h >> 3;
    size_t t0 = (size_t)b * SEQ + (size_t)c * CHUNKL;

    __shared__ float acum[64];
    __shared__ __bf16 sC[64*66], sB[64*66], sX[64*66], sXT[64*66], sM[64*66];

    if (threadIdx.x == 0){
        float s = 0.f;
        for (int i = 0; i < 64; ++i){ s += dAa[(t0+i)*NH + h]; acum[i] = s; }
    }
    __syncthreads();
    if (threadIdx.x < 64)
        gAcum[(((size_t)(b*NH + h))*NC + c)*64 + threadIdx.x] = acum[threadIdx.x];
    float alast = acum[63];

#pragma unroll
    for (int i = 0; i < 16; ++i){
        int e = threadIdx.x + i*256;
        int s = e >> 6, j = e & 63;
        const float* xr = xconv + (t0 + s) * CDIM;
        sC[s*66 + j] = (__bf16)xr[2*DXB + h*HD + j];
        sB[s*66 + j] = (__bf16)xr[DXB   + g*HD + j];
        float dtv = dtsp[(t0+s)*NH + h];
        float xd  = xr[g*HD + j] * dtv;
        sX[s*66 + j]  = (__bf16)xd;
        sXT[j*66 + s] = (__bf16)(xd * __expf(alast - acum[s]));
    }
    __syncthreads();

    int w = threadIdx.x >> 5, lane = threadIdx.x & 31;
    int wm = (w & 3) * 16, wn = (w >> 2) * 32;

    // ---- G = C @ B^T, mask with segsum decays (branchless), write M to LDS
    {
        v8f acc[2] = {};
#pragma unroll
        for (int kk = 0; kk < 64; kk += 32){
            v16bf a = load_A_frag(sC + wm*66 + kk, 66);
#pragma unroll
            for (int ni = 0; ni < 2; ++ni){
                v16bf bb = load_Bt_frag(sB + (wn + ni*16)*66 + kk, 66);
                acc[ni] = wmma_bf(a, bb, acc[ni]);
            }
        }
#pragma unroll
        for (int ni = 0; ni < 2; ++ni)
#pragma unroll
            for (int v = 0; v < 8; ++v){
                int r  = wm + v + ((lane >> 4) << 3);
                int cc = wn + ni*16 + (lane & 15);
                float dec = __expf(acum[r] - acum[cc]);
                float val = acc[ni][v] * dec;
                val = (r >= cc) ? val : 0.f;
                sM[r*66 + cc] = (__bf16)val;
            }
    }
    __syncthreads();

    // ---- Y_diag = M @ Xd
    {
        v8f acc[2] = {};
#pragma unroll
        for (int kk = 0; kk < 64; kk += 32){
            v16bf a = load_A_frag(sM + wm*66 + kk, 66);
#pragma unroll
            for (int ni = 0; ni < 2; ++ni){
                v16bf bb = load_B_frag(sX + kk*66 + wn + ni*16, 66);
                acc[ni] = wmma_bf(a, bb, acc[ni]);
            }
        }
#pragma unroll
        for (int ni = 0; ni < 2; ++ni)
#pragma unroll
            for (int v = 0; v < 8; ++v){
                int r  = wm + v + ((lane >> 4) << 3);
                int cc = wn + ni*16 + (lane & 15);
                gY[(t0 + r)*DI + h*HD + cc] = acc[ni][v];
            }
    }

    // ---- states = (decay.*Xd)^T @ B  -> (p,n)
    {
        v8f acc[2] = {};
#pragma unroll
        for (int kk = 0; kk < 64; kk += 32){
            v16bf a = load_A_frag(sXT + wm*66 + kk, 66);
#pragma unroll
            for (int ni = 0; ni < 2; ++ni){
                v16bf bb = load_B_frag(sB + kk*66 + wn + ni*16, 66);
                acc[ni] = wmma_bf(a, bb, acc[ni]);
            }
        }
        size_t base = ((((size_t)b*NC + c)*NH + h)*HD)*DSTATE;
#pragma unroll
        for (int ni = 0; ni < 2; ++ni)
#pragma unroll
            for (int v = 0; v < 8; ++v){
                int r  = wm + v + ((lane >> 4) << 3);
                int cc = wn + ni*16 + (lane & 15);
                gStates[base + (size_t)r*DSTATE + cc] = acc[ni][v];
            }
    }
}

// ---------------------------------------------------------------- inter-chunk state scan
__global__ __launch_bounds__(256) void hdl_scan(const float* __restrict__ gAcum,
                                                const float* __restrict__ gStates,
                                                float* __restrict__ gPrev){
    int bh = blockIdx.x;
    int b = bh >> 6, h = bh & 63;
    float carry[16];
#pragma unroll
    for (int i = 0; i < 16; ++i) carry[i] = 0.f;
    for (int c = 0; c < NC; ++c){
        float cd = __expf(gAcum[(size_t)bh*NC*64 + (size_t)c*64 + 63]);
        size_t base = ((((size_t)b*NC + c)*NH + h)*HD)*DSTATE;
#pragma unroll
        for (int i = 0; i < 16; ++i){
            size_t e = (size_t)threadIdx.x + (size_t)i*256;
            gPrev[base + e] = carry[i];
            carry[i] = cd * carry[i] + gStates[base + e];
        }
    }
}

// ---------------------------------------------------------------- Y += (C @ prev^T)*out_decay + D*X
__global__ __launch_bounds__(256) void hdl_chunk_off(const float* __restrict__ xconv,
                                                     const float* __restrict__ gAcum,
                                                     const float* __restrict__ gPrev,
                                                     const float* __restrict__ Dv,
                                                     float* __restrict__ gY){
    int blk = blockIdx.x;
    int h = blk & 63, c = (blk >> 6) & 63, b = blk >> 12;
    int g = h >> 3;
    size_t t0 = (size_t)b * SEQ + (size_t)c * CHUNKL;

    __shared__ float acum[64];
    __shared__ __bf16 sC[64*66], sP[64*66];
    if (threadIdx.x < 64)
        acum[threadIdx.x] = gAcum[(((size_t)(b*NH + h))*NC + c)*64 + threadIdx.x];

    size_t pbase = ((((size_t)b*NC + c)*NH + h)*HD)*DSTATE;
#pragma unroll
    for (int i = 0; i < 16; ++i){
        int e = threadIdx.x + i*256;
        int s = e >> 6, j = e & 63;
        sC[s*66 + j] = (__bf16)xconv[(t0 + s)*CDIM + 2*DXB + h*HD + j];
        sP[s*66 + j] = (__bf16)gPrev[pbase + (size_t)s*DSTATE + j];
    }
    __syncthreads();

    int w = threadIdx.x >> 5, lane = threadIdx.x & 31;
    int wm = (w & 3) * 16, wn = (w >> 2) * 32;
    v8f acc[2] = {};
#pragma unroll
    for (int kk = 0; kk < 64; kk += 32){
        v16bf a = load_A_frag(sC + wm*66 + kk, 66);
#pragma unroll
        for (int ni = 0; ni < 2; ++ni){
            v16bf bb = load_Bt_frag(sP + (wn + ni*16)*66 + kk, 66);
            acc[ni] = wmma_bf(a, bb, acc[ni]);
        }
    }
    float dh = Dv[h];
#pragma unroll
    for (int ni = 0; ni < 2; ++ni)
#pragma unroll
        for (int v = 0; v < 8; ++v){
            int r  = wm + v + ((lane >> 4) << 3);
            int cc = wn + ni*16 + (lane & 15);
            float od = __expf(acum[r]);
            size_t yi = (t0 + r)*DI + h*HD + cc;
            float xv = xconv[(t0 + r)*CDIM + g*HD + cc];
            gY[yi] = gY[yi] + acc[ni][v]*od + dh*xv;
        }
}

// ---------------------------------------------------------------- silu(g)*u -> bf16 hi/lo
__global__ __launch_bounds__(256) void hdl_swiglu_split(const float* __restrict__ g,
                                                        const float* __restrict__ u,
                                                        __bf16* __restrict__ hi,
                                                        __bf16* __restrict__ lo, size_t n){
    size_t i = (size_t)blockIdx.x * 256 + threadIdx.x;
    if (i < n){
        float v = siluf(g[i]) * u[i];
        __bf16 h = (__bf16)v;
        hi[i] = h;
        lo[i] = (__bf16)(v - (float)h);
    }
}

// ---------------------------------------------------------------- split-precision bf16 WMMA GEMM
// Double-buffered GLOBAL_LOAD_ASYNC_TO_LDS_B128 staging, s_wait_asynccnt pipelining.
// LDS element-offset layout (per stage, 4096 elems each):
//   Ah: [0|4096]  Al: [8192|12288]  Bh: [16384|20480]  Bl: [24576|28672]  -> 64 KB total
__global__ __launch_bounds__(256) void hdl_gemm_split(const __bf16* __restrict__ Ahi,
                                                      const __bf16* __restrict__ Alo,
                                                      const __bf16* __restrict__ Bhi,
                                                      const __bf16* __restrict__ Blo,
                                                      const float* __restrict__ resid,
                                                      float* __restrict__ C,
                                                      int M, int N, int K){
    __shared__ __align__(128) __bf16 smem[32768];
    const uint32_t lds0 = (uint32_t)(uintptr_t)smem;
    int m0 = blockIdx.y * 128, n0 = blockIdx.x * 128;
    int tid = threadIdx.x;
    int w = tid >> 5, lane = tid & 31;
    int wm = (w & 3) * 32, wn = (w >> 2) * 64;
    int nk = K >> 5;

    // per-thread DMA chunk coordinates (16B = 8 bf16 each; 8 issues per thread per stage)
    int arow = tid >> 2, apart = tid & 3;     // A: 128x32, 4 chunks/row
    int brow = tid >> 4, bpart = tid & 15;    // B: 32x128, 16 chunks/row

    auto issue_tile = [&](int k0, int stage){
        uint32_t sA = lds0 + (uint32_t)(stage * 4096 * 2);
        uint32_t sB = lds0 + (uint32_t)((16384 + stage * 4096) * 2);
#pragma unroll
        for (int half = 0; half < 2; ++half){
            int r = arow + half * 64;
            async_ld16(sA + (uint32_t)(r*64 + apart*16),
                       Ahi + (size_t)(m0 + r)*K + k0 + apart*8);
            async_ld16(sA + (uint32_t)(8192*2 + r*64 + apart*16),
                       Alo + (size_t)(m0 + r)*K + k0 + apart*8);
        }
#pragma unroll
        for (int half = 0; half < 2; ++half){
            int r = brow + half * 16;
            async_ld16(sB + (uint32_t)(r*256 + bpart*16),
                       Bhi + (size_t)(k0 + r)*N + n0 + bpart*8);
            async_ld16(sB + (uint32_t)(8192*2 + r*256 + bpart*16),
                       Blo + (size_t)(k0 + r)*N + n0 + bpart*8);
        }
    };

    issue_tile(0, 0);
    v8f acc[2][4] = {};

    for (int it = 0; it < nk; ++it){
        int stage = it & 1;
        if (it + 1 < nk){
            __builtin_prefetch(Bhi + (size_t)((it + 2) << 5) * N + n0, 0, 1);
            issue_tile((it + 1) << 5, stage ^ 1);
            wait_async8();      // in-order completion => stage `it` fully landed
        } else {
            wait_async0();
        }
        __syncthreads();

        const __bf16* sAh = smem + stage * 4096;
        const __bf16* sAl = smem + 8192  + stage * 4096;
        const __bf16* sBh = smem + 16384 + stage * 4096;
        const __bf16* sBl = smem + 24576 + stage * 4096;
#pragma unroll
        for (int mi = 0; mi < 2; ++mi){
            v16bf ah = load_A_frag(sAh + (wm + mi*16)*32, 32);
            v16bf al = load_A_frag(sAl + (wm + mi*16)*32, 32);
#pragma unroll
            for (int ni = 0; ni < 4; ++ni){
                v16bf bh = load_B_frag(sBh + wn + ni*16, 128);
                v16bf bl = load_B_frag(sBl + wn + ni*16, 128);
                v8f a0 = acc[mi][ni];
                a0 = wmma_bf(ah, bh, a0);
                a0 = wmma_bf(ah, bl, a0);
                a0 = wmma_bf(al, bh, a0);
                acc[mi][ni] = a0;
            }
        }
        __syncthreads();   // stage buffer reused two iterations later
    }

#pragma unroll
    for (int mi = 0; mi < 2; ++mi)
#pragma unroll
        for (int ni = 0; ni < 4; ++ni)
#pragma unroll
            for (int v = 0; v < 8; ++v){
                int r  = m0 + wm + mi*16 + v + ((lane >> 4) << 3);
                int cc = n0 + wn + ni*16 + (lane & 15);
                if (cc < N){
                    float val = acc[mi][ni][v];
                    if (resid) val += resid[(size_t)r*N + cc];
                    C[(size_t)r*N + cc] = val;
                }
            }
}

// ---------------------------------------------------------------- host launcher
extern "C" void kernel_launch(void* const* d_in, const int* in_sizes, int n_in,
                              void* d_out, int out_size, void* d_ws, size_t ws_size,
                              hipStream_t stream){
    (void)in_sizes; (void)n_in; (void)out_size; (void)ws_size;
    const float* hidden = (const float*)d_in[0];
    const float* w_in   = (const float*)d_in[1];
    const float* conv_w = (const float*)d_in[2];
    const float* conv_b = (const float*)d_in[3];
    const float* dt_b   = (const float*)d_in[4];
    const float* A_log  = (const float*)d_in[5];
    const float* Dv     = (const float*)d_in[6];
    const float* norm_w = (const float*)d_in[7];
    const float* w_out  = (const float*)d_in[8];
    const float* ln1    = (const float*)d_in[9];
    const float* ln2    = (const float*)d_in[10];
    const float* gate_w = (const float*)d_in[11];
    const float* up_w   = (const float*)d_in[12];
    const float* down_w = (const float*)d_in[13];

    char* W = (char*)d_ws;
    size_t o = 0;
    auto take = [&](size_t bytes){ size_t r = o; o = (o + bytes + 255) & ~(size_t)255; return r; };

    const size_t szWin = (size_t)DM*NZ, szWout = (size_t)DI*DM;
    const size_t szG = (size_t)DM*INTER, szDn = (size_t)INTER*DM;

    __bf16* WiH = (__bf16*)(W + take(szWin*2));  __bf16* WiL = (__bf16*)(W + take(szWin*2));
    __bf16* WoH = (__bf16*)(W + take(szWout*2)); __bf16* WoL = (__bf16*)(W + take(szWout*2));
    __bf16* GwH = (__bf16*)(W + take(szG*2));    __bf16* GwL = (__bf16*)(W + take(szG*2));
    __bf16* UwH = (__bf16*)(W + take(szG*2));    __bf16* UwL = (__bf16*)(W + take(szG*2));
    __bf16* DwH = (__bf16*)(W + take(szDn*2));   __bf16* DwL = (__bf16*)(W + take(szDn*2));

    __bf16* HnH = (__bf16*)(W + take((size_t)TOK*DM*2));
    __bf16* HnL = (__bf16*)(W + take((size_t)TOK*DM*2));
    float*  Hbuf = (float*)(W + take((size_t)TOK*DM*4));
    __bf16* H2H = (__bf16*)(W + take((size_t)TOK*DM*2));
    __bf16* H2L = (__bf16*)(W + take((size_t)TOK*DM*2));

    size_t scratch0 = o;                                   // Mamba scratch (reused by MLP)
    float* Zx    = (float*)(W + take((size_t)TOK*NZ*4));
    float* Xc    = (float*)(W + take((size_t)TOK*CDIM*4));
    float* DtSp  = (float*)(W + take((size_t)TOK*NH*4));
    float* DAa   = (float*)(W + take((size_t)TOK*NH*4));
    float* Acum  = (float*)(W + take((size_t)BATCH*NH*NC*64*4));
    float* States= (float*)(W + take((size_t)BATCH*NC*NH*HD*DSTATE*4));
    float* Prev  = (float*)(W + take((size_t)BATCH*NC*NH*HD*DSTATE*4));
    float* Yb    = (float*)(W + take((size_t)TOK*DI*4));
    __bf16* UH   = (__bf16*)(W + take((size_t)TOK*DI*2));
    __bf16* UL   = (__bf16*)(W + take((size_t)TOK*DI*2));

    size_t mlpSz = ((size_t)TOK*INTER*4 + 255) & ~(size_t)255;
    float*  Gb = (float*)(W + scratch0);
    float*  Ub = (float*)(W + scratch0 + mlpSz);
    __bf16* AH = (__bf16*)(W + scratch0 + 2*mlpSz);
    __bf16* AL = (__bf16*)(W + scratch0 + 2*mlpSz + (((size_t)TOK*INTER*2 + 255) & ~(size_t)255));

    // 1) weight splits
    hdl_split<<<(unsigned)((szWin +255)/256),256,0,stream>>>(w_in,   WiH, WiL, szWin);
    hdl_split<<<(unsigned)((szWout+255)/256),256,0,stream>>>(w_out,  WoH, WoL, szWout);
    hdl_split<<<(unsigned)((szG  +255)/256),256,0,stream>>>(gate_w, GwH, GwL, szG);
    hdl_split<<<(unsigned)((szG  +255)/256),256,0,stream>>>(up_w,   UwH, UwL, szG);
    hdl_split<<<(unsigned)((szDn +255)/256),256,0,stream>>>(down_w, DwH, DwL, szDn);

    // 2) Mamba branch
    hdl_rmsnorm_split<<<TOK,256,0,stream>>>(hidden, ln1, HnH, HnL, DM);
    hdl_gemm_split<<<dim3((NZ+127)/128, TOK/128),256,0,stream>>>(HnH,HnL, WiH,WiL, nullptr, Zx, TOK, NZ, DM);
    hdl_conv_silu<<<(unsigned)(((size_t)TOK*CDIM+255)/256),256,0,stream>>>(Zx, conv_w, conv_b, Xc);
    hdl_dtprep<<<(unsigned)(((size_t)TOK*NH+255)/256),256,0,stream>>>(Zx, dt_b, A_log, DtSp, DAa);
    hdl_chunk_fwd<<<BATCH*NC*NH,256,0,stream>>>(Xc, DtSp, DAa, Acum, States, Yb);
    hdl_scan<<<BATCH*NH,256,0,stream>>>(Acum, States, Prev);
    hdl_chunk_off<<<BATCH*NC*NH,256,0,stream>>>(Xc, Acum, Prev, Dv, Yb);
    hdl_gatednorm_split<<<TOK,256,0,stream>>>(Yb, Zx, norm_w, UH, UL);
    hdl_gemm_split<<<dim3(DM/128, TOK/128),256,0,stream>>>(UH,UL, WoH,WoL, hidden, Hbuf, TOK, DM, DI);

    // 3) MLP branch
    hdl_rmsnorm_split<<<TOK,256,0,stream>>>(Hbuf, ln2, H2H, H2L, DM);
    hdl_gemm_split<<<dim3(INTER/128, TOK/128),256,0,stream>>>(H2H,H2L, GwH,GwL, nullptr, Gb, TOK, INTER, DM);
    hdl_gemm_split<<<dim3(INTER/128, TOK/128),256,0,stream>>>(H2H,H2L, UwH,UwL, nullptr, Ub, TOK, INTER, DM);
    hdl_swiglu_split<<<(unsigned)(((size_t)TOK*INTER+255)/256),256,0,stream>>>(Gb, Ub, AH, AL, (size_t)TOK*INTER);
    hdl_gemm_split<<<dim3(DM/128, TOK/128),256,0,stream>>>(AH,AL, DwH,DwL, Hbuf, (float*)d_out, TOK, DM, INTER);
}
